// MultiHeadAttention_11690900980293
// MI455X (gfx1250) — compile-verified, matrix operand order checked by reading, not run on hardware
//
#include <hip/hip_runtime.h>

// MultiHeadAttention forward for MI455X (gfx1250), wave32, WMMA f16->f32.
// B=2, S=2048, D=1024, H=16, DH=64.
//
//  1) qkv_proj: x @ W^T + b per head, f16 WMMA, results to d_ws as f16
//     in [B,H,S,DH] layout (Q,K,V = 3 * 8MB = 24MB of workspace).
//  2) flash_attn: online-softmax flash attention; KV tiles of 64 rows,
//     16 WMMAs per iteration; K tiles staged in LDS via
//     global_load_async_to_lds_b128 (ASYNCcnt), V staged transposed.

#define BB 2
#define SS 2048
#define DD 1024
#define HH 16
#define DH 64

typedef __attribute__((ext_vector_type(16))) _Float16 v16h;
typedef __attribute__((ext_vector_type(8)))  float    v8f;

__device__ __forceinline__ v8f wmma16(v16h a, v16h b, v8f c) {
    // D = A(16x32 f16) * B(32x16 f16) + C(16x16 f32)
    return __builtin_amdgcn_wmma_f32_16x16x32_f16(
        /*neg_a=*/false, a, /*neg_b=*/false, b,
        /*c_mod=*/(short)0, c, /*reuse_a=*/false, /*reuse_b=*/false);
}

// Load one 16-half fragment per lane as two aligned 16B chunks.
// rowp points at the lane's row start (+ kchunk*32 already applied);
// kbase = 8*(lane>=16). Matches ISA 16-bit A/B VGPR striping.
__device__ __forceinline__ v16h load_frag(const _Float16* rowp, int kbase) {
    union { v16h h; uint4 u[2]; } f;
    f.u[0] = *(const uint4*)(rowp + kbase);
    f.u[1] = *(const uint4*)(rowp + 16 + kbase);
    return f.h;
}

// ---------------------------------------------------------------------------
// Kernel 1: per-head QKV projection.  grid = (S/128, H, B), block = 256 (8 waves)
// ---------------------------------------------------------------------------
__global__ __launch_bounds__(256) void qkv_proj_kernel(
    const float* __restrict__ seqs,
    const float* __restrict__ Wq, const float* __restrict__ bq,
    const float* __restrict__ Wk, const float* __restrict__ bk,
    const float* __restrict__ Wv, const float* __restrict__ bv,
    _Float16* __restrict__ Q, _Float16* __restrict__ K, _Float16* __restrict__ V)
{
    const int tile = blockIdx.x;
    const int h    = blockIdx.y;
    const int b    = blockIdx.z;
    const int tid  = threadIdx.x;
    const int lane = tid & 31;
    const int wid  = tid >> 5;
    const int mn   = lane & 15;            // M (A row) or N (B/C col) index
    const int kb   = (lane >> 4) << 3;     // 0 or 8: K-half / C-row offset

    __shared__ __align__(16) _Float16 x_lds[128][DH];      // 16 KB
    __shared__ __align__(16) _Float16 w_lds[3][DH][DH];    // 24 KB

    const int row0 = tile * 128;

    // Stage x slice for this head (f32 -> f16), coalesced.
    for (int i = tid; i < 128 * DH; i += 256) {
        const int r = i >> 6, c = i & 63;
        x_lds[r][c] = (_Float16)seqs[((size_t)b * SS + row0 + r) * DD + h * DH + c];
    }
    // Stage the three weight matrices row-major [e][d] (B[k=d][n=e] = W[e][d]).
    const float* Ws[3] = { Wq, Wk, Wv };
#pragma unroll
    for (int j = 0; j < 3; ++j) {
        const float* w = Ws[j] + h * DH * DH;
        for (int i = tid; i < DH * DH; i += 256)
            w_lds[j][i >> 6][i & 63] = (_Float16)w[i];
    }
    __syncthreads();

    // A fragments: x tile rows [wid*16, wid*16+16), K = DH split in 2 chunks.
    v16h a[2];
#pragma unroll
    for (int kc = 0; kc < 2; ++kc)
        a[kc] = load_frag(&x_lds[wid * 16 + mn][kc * 32], kb);

    const float* biases[3] = { bq, bk, bv };
    _Float16*    outs[3]   = { Q, K, V };
    const size_t obase = ((size_t)(b * HH + h) * SS + row0 + wid * 16) * DH;

#pragma unroll
    for (int j = 0; j < 3; ++j) {
#pragma unroll
        for (int nt = 0; nt < 4; ++nt) {
            v8f acc = {};
#pragma unroll
            for (int kc = 0; kc < 2; ++kc) {
                v16h bf = load_frag(&w_lds[j][nt * 16 + mn][kc * 32], kb);
                acc = wmma16(a[kc], bf, acc);
            }
            const float bias = biases[j][h * DH + nt * 16 + mn];
            _Float16* o = outs[j] + obase;
#pragma unroll
            for (int r = 0; r < 8; ++r) {
                const int mrow = r + kb;                       // C layout row
                o[(size_t)mrow * DH + nt * 16 + mn] = (_Float16)(acc[r] + bias);
            }
        }
    }
}

// ---------------------------------------------------------------------------
// Kernel 2: flash attention.  grid = (S/128, H, B), block = 256 (8 waves);
// each wave owns 16 query rows, loop over S in KV tiles of 64 rows.
// ---------------------------------------------------------------------------
__global__ __launch_bounds__(256) void flash_attn_kernel(
    const _Float16* __restrict__ Q, const _Float16* __restrict__ K,
    const _Float16* __restrict__ V, float* __restrict__ out)
{
    const int qt   = blockIdx.x;
    const int h    = blockIdx.y;
    const int b    = blockIdx.z;
    const int tid  = threadIdx.x;
    const int lane = tid & 31;
    const int wid  = tid >> 5;
    const int mn   = lane & 15;
    const int kb   = (lane >> 4) << 3;

    __shared__ __align__(16) _Float16 k_lds[64][DH];        // 8 KB (async target)
    __shared__ __align__(16) _Float16 vt_lds[DH][64];       // 8 KB (V transposed)
    __shared__ __align__(16) _Float16 p_lds[8][16][64];     // 16 KB (per-wave P)

    const size_t head_base = (size_t)(b * HH + h) * SS * DH;
    const _Float16* Qh = Q + head_base;
    const _Float16* Kh = K + head_base;
    const _Float16* Vh = V + head_base;

    // Q A-fragments straight from global (row-major [s][d], 16B aligned).
    const int q0 = qt * 128 + wid * 16;
    v16h aq[2];
#pragma unroll
    for (int kc = 0; kc < 2; ++kc)
        aq[kc] = load_frag(Qh + (size_t)(q0 + mn) * DH + kc * 32, kb);

    v8f o0 = {}, o1 = {}, o2 = {}, o3 = {};
    float mrow[8], lrow[8];
#pragma unroll
    for (int r = 0; r < 8; ++r) { mrow[r] = -1e30f; lrow[r] = 0.0f; }

    const unsigned klds_base = (unsigned)(size_t)&k_lds[0][0];
    const float scale = 0.125f;   // 1/sqrt(64)

    for (int jt = 0; jt < SS / 64; ++jt) {
        const _Float16* kg = Kh + (size_t)jt * 64 * DH;
        const _Float16* vg = Vh + (size_t)jt * 64 * DH;

        if (jt + 1 < SS / 64) {                 // pull next tiles toward L2/L0
            __builtin_prefetch(kg + 64 * DH, 0, 0);
            __builtin_prefetch(vg + 64 * DH, 0, 0);
        }

        // K tile: 64x64 f16 = 8192 B = 256 threads x 32 B, async copy to LDS.
        {
            const unsigned off = (unsigned)tid * 32u;
            const unsigned lds_addr = klds_base + off;
            const unsigned long long ga = (unsigned long long)(size_t)kg + off;
            asm volatile("global_load_async_to_lds_b128 %0, %1, off"
                         :: "v"(lds_addr), "v"(ga) : "memory");
            asm volatile("global_load_async_to_lds_b128 %0, %1, off offset:16"
                         :: "v"(lds_addr), "v"(ga) : "memory");
        }
        // V tile: stage transposed (vt[d][kvrow]) so PV B-fragments are rows.
        for (int i = tid; i < 64 * DH; i += 256) {
            const int r = i >> 6, c = i & 63;
            vt_lds[c][r] = vg[i];
        }
        asm volatile("s_wait_asynccnt 0x0" ::: "memory");
        __syncthreads();

        // scores[16x64] = Q(16x64) @ K^T : B[k=d][n=kvrow] = k_lds[kvrow][d]
        v8f s0 = {}, s1 = {}, s2 = {}, s3 = {};
#pragma unroll
        for (int kc = 0; kc < 2; ++kc) {
            s0 = wmma16(aq[kc], load_frag(&k_lds[ 0 + mn][kc * 32], kb), s0);
            s1 = wmma16(aq[kc], load_frag(&k_lds[16 + mn][kc * 32], kb), s1);
            s2 = wmma16(aq[kc], load_frag(&k_lds[32 + mn][kc * 32], kb), s2);
            s3 = wmma16(aq[kc], load_frag(&k_lds[48 + mn][kc * 32], kb), s3);
        }

        // Online softmax. Row m spans 16 lanes (n) at fixed vgpr r; reduce
        // within width-16 so the two row-halves stay independent.
        float alpha[8];
#pragma unroll
        for (int r = 0; r < 8; ++r) {
            const float x0 = s0[r] * scale, x1 = s1[r] * scale;
            const float x2 = s2[r] * scale, x3 = s3[r] * scale;
            float mx = fmaxf(fmaxf(x0, x1), fmaxf(x2, x3));
#pragma unroll
            for (int o = 1; o < 16; o <<= 1) mx = fmaxf(mx, __shfl_xor(mx, o, 16));
            const float mnew = fmaxf(mrow[r], mx);
            const float a    = __expf(mrow[r] - mnew);
            const float p0   = __expf(x0 - mnew);
            const float p1   = __expf(x1 - mnew);
            const float p2   = __expf(x2 - mnew);
            const float p3   = __expf(x3 - mnew);
            float rs = (p0 + p1) + (p2 + p3);
#pragma unroll
            for (int o = 1; o < 16; o <<= 1) rs += __shfl_xor(rs, o, 16);
            lrow[r] = lrow[r] * a + rs;
            mrow[r] = mnew;
            alpha[r] = a;
            // Re-layout P (C layout -> LDS row-major [m][n]) for A-fragment use.
            const int m = r + kb;
            p_lds[wid][m][ 0 + mn] = (_Float16)p0;
            p_lds[wid][m][16 + mn] = (_Float16)p1;
            p_lds[wid][m][32 + mn] = (_Float16)p2;
            p_lds[wid][m][48 + mn] = (_Float16)p3;
        }
#pragma unroll
        for (int r = 0; r < 8; ++r) {
            o0[r] *= alpha[r]; o1[r] *= alpha[r];
            o2[r] *= alpha[r]; o3[r] *= alpha[r];
        }
        asm volatile("" ::: "memory");  // order p_lds stores before reloads

        // O += P(16x64) @ V(64x64): B[k=kvrow][n=d] = vt_lds[d][kvrow]
        v16h ap[2];
#pragma unroll
        for (int kc = 0; kc < 2; ++kc)
            ap[kc] = load_frag(&p_lds[wid][mn][kc * 32], kb);
#pragma unroll
        for (int kc = 0; kc < 2; ++kc) {
            o0 = wmma16(ap[kc], load_frag(&vt_lds[0 * 16 + mn][kc * 32], kb), o0);
            o1 = wmma16(ap[kc], load_frag(&vt_lds[1 * 16 + mn][kc * 32], kb), o1);
            o2 = wmma16(ap[kc], load_frag(&vt_lds[2 * 16 + mn][kc * 32], kb), o2);
            o3 = wmma16(ap[kc], load_frag(&vt_lds[3 * 16 + mn][kc * 32], kb), o3);
        }

        __syncthreads();   // protect k_lds / vt_lds before next tile
    }

    // Epilogue: normalize and scatter f32 output [B,S,D] (D = H*DH).
    float inv[8];
#pragma unroll
    for (int r = 0; r < 8; ++r) inv[r] = 1.0f / lrow[r];

    const size_t obase = (size_t)b * SS + q0;
#pragma unroll
    for (int r = 0; r < 8; ++r) {
        const size_t row = obase + (size_t)(r + kb);
        float* orow = out + row * DD + h * DH;
        orow[0 * 16 + mn] = o0[r] * inv[r];
        orow[1 * 16 + mn] = o1[r] * inv[r];
        orow[2 * 16 + mn] = o2[r] * inv[r];
        orow[3 * 16 + mn] = o3[r] * inv[r];
    }
}

// ---------------------------------------------------------------------------
extern "C" void kernel_launch(void* const* d_in, const int* in_sizes, int n_in,
                              void* d_out, int out_size, void* d_ws, size_t ws_size,
                              hipStream_t stream) {
    const float* seqs = (const float*)d_in[0];
    const float* Wq   = (const float*)d_in[1];
    const float* bq   = (const float*)d_in[2];
    const float* Wk   = (const float*)d_in[3];
    const float* bk   = (const float*)d_in[4];
    const float* Wv   = (const float*)d_in[5];
    const float* bv   = (const float*)d_in[6];
    float* out = (float*)d_out;

    // Workspace: Q,K,V f16 [B,H,S,DH] -> 3 * 8 MB = 24 MB.
    const size_t elems = (size_t)BB * HH * SS * DH;
    _Float16* Qw = (_Float16*)d_ws;
    _Float16* Kw = Qw + elems;
    _Float16* Vw = Kw + elems;

    dim3 grid(SS / 128, HH, BB), block(256);
    qkv_proj_kernel<<<grid, block, 0, stream>>>(seqs, Wq, bq, Wk, bk, Wv, bv,
                                                Qw, Kw, Vw);
    flash_attn_kernel<<<grid, block, 0, stream>>>(Qw, Kw, Vw, out);
}
